// MHSARPB_35442070127173
// MI455X (gfx1250) — matrix operand order repaired and checked by reading, not compiled, not run on hardware
//
#include <hip/hip_runtime.h>
#include <hip/hip_bf16.h>

typedef __attribute__((ext_vector_type(16))) __bf16          v16bf;
typedef __attribute__((ext_vector_type(8)))  float           v8f;

#define WMMA_BF16(a, b, c) \
  __builtin_amdgcn_wmma_f32_16x16x32_bf16(false, (a), false, (b), (short)0, (c), false, false)

// ---- zero-VALU fragment assembly: two b128 loads bit-cast straight to v16bf
struct Pair128 { uint4 lo, hi; };
static __device__ __forceinline__ v16bf frag_from(uint4 lo, uint4 hi) {
  Pair128 t{lo, hi};
  return __builtin_bit_cast(v16bf, t);
}

// native fp32 -> bf16
static __device__ __forceinline__ unsigned short bfbits(float f) {
  __bf16 h = (__bf16)f;
  return __builtin_bit_cast(unsigned short, h);
}

// A fragment (16x32 bf16, MxK), bf16 row-major source.
static __device__ __forceinline__ v16bf loadA_bf16(const unsigned short* base, int ld,
                                                   int row0, int k0, int lane) {
  const unsigned short* p = base + (row0 + (lane & 15)) * ld + k0 + ((lane >> 4) << 4);
  return frag_from(*(const uint4*)p, *(const uint4*)(p + 8));
}

// B fragment (32x16 bf16, KxN) where B(k,n) = W[n][k], W bf16 row-major.
static __device__ __forceinline__ v16bf loadB_bf16(const unsigned short* W, int ld,
                                                   int n0, int k0, int lane) {
  const unsigned short* p = W + (n0 + (lane & 15)) * ld + k0 + ((lane >> 4) << 4);
  return frag_from(((const uint4*)p)[0], ((const uint4*)p)[1]);
}

// B fragment from 16 contiguous bf16 (caller resolves lane -> pointer).
static __device__ __forceinline__ v16bf loadBrow_bf16(const unsigned short* p) {
  return frag_from(((const uint4*)p)[0], ((const uint4*)p)[1]);
}

// ---------------------------------------------------------------------------
// Kernel 0: fp32 -> bf16 bulk convert (8 elems/thread)
// ---------------------------------------------------------------------------
__global__ __launch_bounds__(256) void cvt_kernel(const float* __restrict__ in,
                                                  unsigned short* __restrict__ out, int n8) {
  int i = blockIdx.x * 256 + threadIdx.x;
  if (i >= n8) return;
  const float4* p = (const float4*)in + (size_t)i * 2;
  float4 f0 = p[0], f1 = p[1];
  uint4 o;
  o.x = (unsigned)bfbits(f0.x) | ((unsigned)bfbits(f0.y) << 16);
  o.y = (unsigned)bfbits(f0.z) | ((unsigned)bfbits(f0.w) << 16);
  o.z = (unsigned)bfbits(f1.x) | ((unsigned)bfbits(f1.y) << 16);
  o.w = (unsigned)bfbits(f1.z) | ((unsigned)bfbits(f1.w) << 16);
  ((uint4*)out)[i] = o;
}

// ---------------------------------------------------------------------------
// Kernel 1: qkv = xb @ wb^T + qkv_b ; scatter q(scaled)/k [b,h,n,d], v^T [b,h,d,n].
// Block 256 thr = 8 waves (4x2), block tile 128x128, wave tile 32x64.
// ---------------------------------------------------------------------------
__global__ __launch_bounds__(256) void qkv_kernel(
    const unsigned short* __restrict__ xb, const unsigned short* __restrict__ wb,
    const float* __restrict__ bias,
    unsigned short* __restrict__ qs, unsigned short* __restrict__ kk,
    unsigned short* __restrict__ vt) {
  const int K = 512, NH = 16, D = 32, N = 1024;
  const int lane = threadIdx.x & 31;
  const int wave = threadIdx.x >> 5;
  const int row0 = blockIdx.x * 128 + (wave & 3) * 32;
  const int col0 = blockIdx.y * 128 + (wave >> 2) * 64;

  v8f acc[2][4] = {};
#pragma unroll 2
  for (int k0 = 0; k0 < K; k0 += 32) {
    v16bf a0 = loadA_bf16(xb, K, row0, k0, lane);
    v16bf a1 = loadA_bf16(xb, K, row0 + 16, k0, lane);
#pragma unroll
    for (int j = 0; j < 4; j++) {
      v16bf bj = loadB_bf16(wb, K, col0 + j * 16, k0, lane);
      acc[0][j] = WMMA_BF16(a0, bj, acc[0][j]);
      acc[1][j] = WMMA_BF16(a1, bj, acc[1][j]);
    }
  }

  const float scale = 0.17677669529663689f;   // 32^-0.5
  const int half8 = (lane >> 4) << 3;
  const int nn = lane & 15;
#pragma unroll
  for (int i = 0; i < 2; i++)
#pragma unroll
    for (int j = 0; j < 4; j++)
#pragma unroll
      for (int r = 0; r < 8; r++) {
        int grow = row0 + i * 16 + r + half8;
        int gcol = col0 + j * 16 + nn;
        float val = acc[i][j][r] + bias[gcol];
        int which = gcol >> 9;
        int rc = gcol & 511;
        int h = rc >> 5, d = rc & 31;
        int b = grow >> 10, n = grow & 1023;
        int bh = b * NH + h;
        if (which == 0)      qs[(bh * N + n) * D + d] = bfbits(val * scale);
        else if (which == 1) kk[(bh * N + n) * D + d] = bfbits(val);
        else                 vt[(bh * D + d) * N + n] = bfbits(val);
      }
}

// ---------------------------------------------------------------------------
// Kernel 2: flash attention per (b,h); one wave = 16-row query band.
// 64 keys/step: 4x QK^T WMMA -> shared rel-pos bias index -> one online
// softmax reduction -> exp(P) via LDS reshape -> 4x PV WMMA.
// Prefetches next K/V tiles (global_prefetch_b8).
// ---------------------------------------------------------------------------
__global__ __launch_bounds__(128) void attn_kernel(
    const unsigned short* __restrict__ qs, const unsigned short* __restrict__ kk,
    const unsigned short* __restrict__ vt, const float* __restrict__ rpb,
    unsigned short* __restrict__ ao) {
  const int N = 1024, D = 32, NH = 16, C = 512, R = 63;
  __shared__ unsigned short P[4][16][64];   // per-wave 16x64 exp(S) tile

  const int lane = threadIdx.x & 31;
  const int wave = threadIdx.x >> 5;
  const int h = blockIdx.y, b = blockIdx.z;
  const int bh = b * NH + h;
  const int m0 = (blockIdx.x * 4 + wave) * 16;

  const unsigned short* qptr = qs + bh * N * D;
  const unsigned short* kptr = kk + bh * N * D;
  const unsigned short* vptr = vt + bh * D * N;
  const float* rp = rpb + h * R * R;

  const int half8 = (lane >> 4) << 3;
  const int half16 = (lane >> 4) << 4;
  const int nn = lane & 15;

  v16bf aq = loadA_bf16(qptr, D, m0, 0, lane);
  v8f acc0 = {}, acc1 = {};
  float mrow[8], lrow[8];
  int qbase[8];   // (yi+31)*63 + xi + 31, loop-invariant per row
#pragma unroll
  for (int r = 0; r < 8; r++) {
    mrow[r] = -1e30f; lrow[r] = 0.0f;
    int ip = (N - 1) - (m0 + r + half8);
    qbase[r] = ((ip >> 5) + 31) * R + (ip & 31) + 31;
  }

  for (int j0 = 0; j0 < N; j0 += 64) {
    if (j0 + 64 < N) {  // scalar-uniform: prefetch next K/V tiles
      __builtin_prefetch(kptr + (j0 + 64 + nn) * D, 0, 3);
      __builtin_prefetch(vptr + nn * N + j0 + 64, 0, 3);
      __builtin_prefetch(vptr + (16 + nn) * N + j0 + 64, 0, 3);
    }

    v8f s[4];
    int kidx[4];    // yj*63 + xj per 16-key subtile (shared by all 8 rows)
#pragma unroll
    for (int t = 0; t < 4; t++) {
      v16bf bk = loadBrow_bf16(kptr + (j0 + t * 16 + nn) * D + half16);
      v8f z = {};
      s[t] = WMMA_BF16(aq, bk, z);
      int jp = (N - 1) - (j0 + t * 16 + nn);
      kidx[t] = (jp >> 5) * R + (jp & 31);
    }

#pragma unroll
    for (int r = 0; r < 8; r++) {
      float a0 = s[0][r] + rp[qbase[r] - kidx[0]];
      float a1 = s[1][r] + rp[qbase[r] - kidx[1]];
      float a2 = s[2][r] + rp[qbase[r] - kidx[2]];
      float a3 = s[3][r] + rp[qbase[r] - kidx[3]];
      float tm = fmaxf(fmaxf(a0, a1), fmaxf(a2, a3));
#pragma unroll
      for (int msk = 1; msk < 16; msk <<= 1) tm = fmaxf(tm, __shfl_xor(tm, msk, 16));
      float mnew = fmaxf(mrow[r], tm);
      float alpha = __expf(mrow[r] - mnew);
      float p0 = __expf(a0 - mnew);
      float p1 = __expf(a1 - mnew);
      float p2 = __expf(a2 - mnew);
      float p3 = __expf(a3 - mnew);
      float ts = (p0 + p1) + (p2 + p3);
#pragma unroll
      for (int msk = 1; msk < 16; msk <<= 1) ts += __shfl_xor(ts, msk, 16);
      lrow[r] = lrow[r] * alpha + ts;
      mrow[r] = mnew;
      acc0[r] *= alpha;
      acc1[r] *= alpha;
      P[wave][r + half8][nn]      = bfbits(p0);
      P[wave][r + half8][nn + 16] = bfbits(p1);
      P[wave][r + half8][nn + 32] = bfbits(p2);
      P[wave][r + half8][nn + 48] = bfbits(p3);
    }
    __syncthreads();   // uniform trip count; orders LDS writes before reads

#pragma unroll
    for (int c = 0; c < 2; c++) {      // two 32-key chunks of P
      unsigned short up[16];
#pragma unroll
      for (int e = 0; e < 8; e++) {
        up[e]     = P[wave][nn][c * 32 + half16 + e];
        up[8 + e] = P[wave][nn][c * 32 + half16 + 8 + e];
      }
      v16bf ap = __builtin_bit_cast(v16bf, up);
      v16bf bv0 = loadBrow_bf16(vptr + nn * N        + j0 + c * 32 + half16);
      v16bf bv1 = loadBrow_bf16(vptr + (16 + nn) * N + j0 + c * 32 + half16);
      acc0 = WMMA_BF16(ap, bv0, acc0);
      acc1 = WMMA_BF16(ap, bv1, acc1);
    }
    __syncthreads();   // reads done before next iteration's writes
  }

#pragma unroll
  for (int r = 0; r < 8; r++) {
    int n = m0 + r + half8;
    float inv = 1.0f / lrow[r];
    int o = (b * N + n) * C + h * D;
    ao[o + nn]      = bfbits(acc0[r] * inv);
    ao[o + 16 + nn] = bfbits(acc1[r] * inv);
  }
}

// ---------------------------------------------------------------------------
// Kernel 3: out = ao @ wprojb^T + proj_b (fp32 out).  Wave tile 32x64.
// ---------------------------------------------------------------------------
__global__ __launch_bounds__(256) void proj_kernel(
    const unsigned short* __restrict__ ao, const unsigned short* __restrict__ wb,
    const float* __restrict__ bias, float* __restrict__ out) {
  const int K = 512, NC = 512;
  const int lane = threadIdx.x & 31;
  const int wave = threadIdx.x >> 5;
  const int row0 = blockIdx.x * 128 + (wave & 3) * 32;
  const int col0 = blockIdx.y * 128 + (wave >> 2) * 64;

  v8f acc[2][4] = {};
#pragma unroll 2
  for (int k0 = 0; k0 < K; k0 += 32) {
    v16bf a0 = loadA_bf16(ao, K, row0, k0, lane);
    v16bf a1 = loadA_bf16(ao, K, row0 + 16, k0, lane);
#pragma unroll
    for (int j = 0; j < 4; j++) {
      v16bf bj = loadB_bf16(wb, K, col0 + j * 16, k0, lane);
      acc[0][j] = WMMA_BF16(a0, bj, acc[0][j]);
      acc[1][j] = WMMA_BF16(a1, bj, acc[1][j]);
    }
  }

  const int half8 = (lane >> 4) << 3;
  const int nn = lane & 15;
#pragma unroll
  for (int i = 0; i < 2; i++)
#pragma unroll
    for (int j = 0; j < 4; j++)
#pragma unroll
      for (int r = 0; r < 8; r++) {
        int grow = row0 + i * 16 + r + half8;
        int gcol = col0 + j * 16 + nn;
        out[grow * NC + gcol] = acc[i][j][r] + bias[gcol];
      }
}

extern "C" void kernel_launch(void* const* d_in, const int* in_sizes, int n_in,
                              void* d_out, int out_size, void* d_ws, size_t ws_size,
                              hipStream_t stream) {
  const float* x      = (const float*)d_in[0];
  const float* qkv_w  = (const float*)d_in[1];
  const float* qkv_b  = (const float*)d_in[2];
  const float* rpb    = (const float*)d_in[3];
  const float* proj_w = (const float*)d_in[4];
  const float* proj_b = (const float*)d_in[5];
  float* out = (float*)d_out;

  const size_t XN  = (size_t)8192 * 512;
  const size_t WQN = (size_t)1536 * 512;
  const size_t WPN = (size_t)512 * 512;
  unsigned short* xb  = (unsigned short*)d_ws;
  unsigned short* wqb = xb  + XN;
  unsigned short* wpb = wqb + WQN;
  unsigned short* qs  = wpb + WPN;
  unsigned short* kk  = qs  + XN;
  unsigned short* vt  = kk  + XN;
  unsigned short* ao  = vt  + XN;

  cvt_kernel<<<dim3((XN / 8 + 255) / 256),  256, 0, stream>>>(x,      xb,  (int)(XN / 8));
  cvt_kernel<<<dim3((WQN / 8 + 255) / 256), 256, 0, stream>>>(qkv_w,  wqb, (int)(WQN / 8));
  cvt_kernel<<<dim3((WPN / 8 + 255) / 256), 256, 0, stream>>>(proj_w, wpb, (int)(WPN / 8));

  qkv_kernel<<<dim3(64, 12), 256, 0, stream>>>(xb, wqb, qkv_b, qs, kk, vt);
  attn_kernel<<<dim3(16, 16, 8), 128, 0, stream>>>(qs, kk, vt, rpb, ao);
  proj_kernel<<<dim3(64, 4), 256, 0, stream>>>(ao, wpb, proj_b, out);
}
